// SparseTransformer_94489281430
// MI455X (gfx1250) — compile-verified, matrix operand order checked by reading, not run on hardware
//
#include <hip/hip_runtime.h>
#include <hip/hip_bf16.h>

// Problem dims
#define BQ 2048
#define NN 768
#define MK 8192
#define EE 512
#define HH 8

typedef __attribute__((ext_vector_type(16))) __bf16 v16bf;
typedef __attribute__((ext_vector_type(8)))  __bf16 v8bf;
typedef __attribute__((ext_vector_type(4)))  __bf16 v4bf;
typedef __attribute__((ext_vector_type(8)))  float  v8f;

#define WMMA_BF16(a, b, c) \
  __builtin_amdgcn_wmma_f32_16x16x32_bf16(false, (a), false, (b), (short)0, (c), false, false)

// LDS byte address (AS3 offset) of a generic pointer to a __shared__ object —
// same value the compiler feeds its own ds_* address VGPRs.
__device__ __forceinline__ unsigned lds_off(const void* p) {
  return (unsigned)(unsigned long long)(const __attribute__((address_space(3))) void*)p;
}

// CDNA5 async copy: 16 bytes global -> LDS per enabled lane, tracked by
// ASYNCcnt (CDNA5 ISA: GLOBAL_LOAD_ASYNC_TO_LDS_B128).
__device__ __forceinline__ void async_ld16(unsigned lds_dst, const void* gsrc) {
  asm volatile("global_load_async_to_lds_b128 %0, %1, off"
               :: "v"(lds_dst), "v"((unsigned long long)gsrc)
               : "memory");
}
__device__ __forceinline__ void wait_async0() {
  asm volatile("s_wait_asynccnt 0x0" ::: "memory");
}

// 16x32 bf16 fragment (A-layout; B-layout symmetric) from row-major LDS.
// ISA 7.12.2: lanes 0-15 hold K=0..7 & 16..23; lanes 16-31 hold K=8..15 & 24..31.
__device__ __forceinline__ v16bf load_frag(const __bf16* base, int row, int ld) {
  const int lane = threadIdx.x & 31;
  const int kb = (lane >> 4) << 3;  // 0 or 8
  const __bf16* p = base + (size_t)row * ld + kb;
  v8bf lo = *(const v8bf*)(p);       // ds_load_b128
  v8bf hi = *(const v8bf*)(p + 16);  // ds_load_b128
  v16bf r;
#pragma unroll
  for (int i = 0; i < 8; ++i) { r[i] = lo[i]; r[8 + i] = hi[i]; }
  return r;
}

// ---------------------------------------------------------------------------
// fp32 -> bf16 conversions (done ONCE so all hot loops stage pure bf16 bytes)
// ---------------------------------------------------------------------------
__global__ __launch_bounds__(256) void cvt_kernel(const float* __restrict__ in,
                                                  __bf16* __restrict__ out, int n4) {
  int i = blockIdx.x * 256 + threadIdx.x;
  if (i < n4) {
    float4 v = ((const float4*)in)[i];
    v4bf r;
    r[0] = (__bf16)v.x; r[1] = (__bf16)v.y; r[2] = (__bf16)v.z; r[3] = (__bf16)v.w;
    ((v4bf*)out)[i] = r;
  }
}

// in: [K,N] fp32 row-major  ->  out: [N,K] bf16 row-major (so GEMM B is always [N,K])
__global__ __launch_bounds__(256) void tcvt_kernel(const float* __restrict__ in,
                                                   __bf16* __restrict__ out, int K, int N) {
  int n = blockIdx.x * 256 + threadIdx.x;
  int k = blockIdx.y;
  out[(size_t)n * K + k] = (__bf16)in[(size_t)k * N + n];
}

// ---------------------------------------------------------------------------
// GEMM: C[M,N] = alpha * A[M,K] @ B[N,K]^T + bias, all-bf16 inputs.
// 256 threads (8 waves), 128x64 block tile, wave grid 4(M)x2(N), 32x32/wave,
// K stepped by 64 -> 8 WMMAs per wave per LDS fill. Tiles staged with
// async-to-LDS. Outputs fp32 (Cf) and/or bf16 (Cb).
// M % 128 == 0, N % 64 == 0, K % 64 == 0 for all calls here.
// ---------------------------------------------------------------------------
__global__ __launch_bounds__(256) void gemm_bf16_kernel(
    const __bf16* __restrict__ A, const __bf16* __restrict__ Bw,
    const float* __restrict__ bias, float* __restrict__ Cf,
    __bf16* __restrict__ Cb, int Ndim, int Kdim, float alpha) {
  __shared__ __align__(16) __bf16 As[128 * 64];  // [row][k]
  __shared__ __align__(16) __bf16 Bs[64 * 64];   // [n][k]
  const int t = threadIdx.x, lane = t & 31, wid = t >> 5, l15 = lane & 15;
  const int wm = wid >> 1, wn = wid & 1;
  const size_t bm = (size_t)blockIdx.y * 128, bn = (size_t)blockIdx.x * 64;

  v8f acc[2][2] = {};
  for (int k0 = 0; k0 < Kdim; k0 += 64) {
    __syncthreads();
#pragma unroll
    for (int j = 0; j < 4; ++j) {  // A tile: 1024 x 16B units
      int u = t + j * 256, r = u >> 3, ko = u & 7;
      async_ld16(lds_off(&As[r * 64 + ko * 8]), A + (bm + r) * Kdim + k0 + ko * 8);
    }
#pragma unroll
    for (int j = 0; j < 2; ++j) {  // B tile: 512 x 16B units
      int u = t + j * 256, r = u >> 3, ko = u & 7;
      async_ld16(lds_off(&Bs[r * 64 + ko * 8]), Bw + (bn + r) * Kdim + k0 + ko * 8);
    }
    wait_async0();
    __syncthreads();
#pragma unroll
    for (int kc = 0; kc < 64; kc += 32) {
      v16bf a0 = load_frag(As + kc, wm * 32 + l15, 64);
      v16bf a1 = load_frag(As + kc, wm * 32 + 16 + l15, 64);
      v16bf b0 = load_frag(Bs + kc, wn * 32 + l15, 64);
      v16bf b1 = load_frag(Bs + kc, wn * 32 + 16 + l15, 64);
      acc[0][0] = WMMA_BF16(a0, b0, acc[0][0]);
      acc[0][1] = WMMA_BF16(a0, b1, acc[0][1]);
      acc[1][0] = WMMA_BF16(a1, b0, acc[1][0]);
      acc[1][1] = WMMA_BF16(a1, b1, acc[1][1]);
    }
  }
#pragma unroll
  for (int cn = 0; cn < 2; ++cn) {
    const size_t col = bn + wn * 32 + cn * 16 + l15;
    const float bvv = bias ? bias[col] : 0.0f;
#pragma unroll
    for (int cm = 0; cm < 2; ++cm)
#pragma unroll
      for (int i = 0; i < 8; ++i) {
        size_t r = bm + wm * 32 + cm * 16 + i + ((lane >> 4) << 3);
        float v = alpha * acc[cm][cn][i] + bvv;
        if (Cf) Cf[r * Ndim + col] = v;
        if (Cb) Cb[r * Ndim + col] = (__bf16)v;
      }
  }
}

// ---------------------------------------------------------------------------
// Flash-style attention (bf16 q/k/v in, bf16 ctx out, fp32 f out).
// 128 blocks x 256 threads; wave h owns head h; block owns 16 query rows.
// Two passes over 8192 keys in 32-key tiles; q/K tiles staged async.
// ---------------------------------------------------------------------------
__global__ __launch_bounds__(256) void attention_kernel(
    const __bf16* __restrict__ qmat,  // [BQ, EE]
    const __bf16* __restrict__ kmat,  // [MK, EE]
    const __bf16* __restrict__ vmat,  // [MK, EE]
    __bf16* __restrict__ ctx,         // [BQ, EE]
    float* __restrict__ fout) {       // [BQ, MK]
  __shared__ __align__(16) __bf16 qs[16 * EE];   // [q][dim]
  __shared__ __align__(16) __bf16 kv[HH][2048];  // K:[key][64] / Vt:[a][32]
  __shared__ __align__(16) __bf16 ps[HH][512];   // p:[q][32 keys]
  const int t = threadIdx.x, lane = t & 31, h = t >> 5, l15 = lane & 15;
  const int b0 = blockIdx.x * 16;
  const float SCALE = 0.125f;  // 1/sqrt(64)

  // q block: async copy 16x512 bf16
#pragma unroll
  for (int j = 0; j < 4; ++j) {
    int u = t + j * 256, r = u >> 6, du = u & 63;
    async_ld16(lds_off(&qs[r * EE + du * 8]), qmat + (size_t)(b0 + r) * EE + du * 8);
  }
  wait_async0();
  __syncthreads();

  float mx[8], sm[8], rs[8];
#pragma unroll
  for (int i = 0; i < 8; ++i) { mx[i] = -1e30f; sm[i] = 0.0f; }
  v8f acc[4] = {};

  for (int pass = 0; pass < 2; ++pass) {
    for (int mt = 0; mt < MK; mt += 32) {
      __syncthreads();
      // K tile: async copy, re-striped per head: kv[h][key*64+d]
#pragma unroll
      for (int j = 0; j < 8; ++j) {
        int u = t + j * 256, key = u >> 6, du = u & 63;
        async_ld16(lds_off(&kv[du >> 3][key * 64 + (du & 7) * 8]),
                   kmat + (size_t)(mt + key) * EE + du * 8);
      }
      wait_async0();
      __syncthreads();

      // scores s[16q x 32keys] for head h
      v8f s0 = {}, s1 = {};
#pragma unroll
      for (int kc = 0; kc < 64; kc += 32) {
        v16bf aq = load_frag(&qs[0] + h * 64 + kc, l15, EE);
        v16bf k0f = load_frag(&kv[h][0] + kc, l15, 64);
        v16bf k1f = load_frag(&kv[h][0] + kc, 16 + l15, 64);
        s0 = WMMA_BF16(aq, k0f, s0);
        s1 = WMMA_BF16(aq, k1f, s1);
      }

      if (pass == 0) {
#pragma unroll
        for (int i = 0; i < 8; ++i) {
          float v0 = s0[i] * SCALE, v1 = s1[i] * SCALE;
          float nm = fmaxf(mx[i], fmaxf(v0, v1));
          sm[i] = sm[i] * __expf(mx[i] - nm) + __expf(v0 - nm) + __expf(v1 - nm);
          mx[i] = nm;
        }
      } else {
#pragma unroll
        for (int i = 0; i < 8; ++i) {
          float p0 = __expf(s0[i] * SCALE - mx[i]) * rs[i];
          float p1 = __expf(s1[i] * SCALE - mx[i]) * rs[i];
          int row = i + ((lane >> 4) << 3);
          ps[h][row * 32 + l15] = (__bf16)p0;
          ps[h][row * 32 + 16 + l15] = (__bf16)p1;
        }
        __syncthreads();  // ps complete; everyone done reading K
        // V tile transposed into same buffer: kv[h][a*32+key]
#pragma unroll
        for (int j = 0; j < 8; ++j) {
          int u = t + j * 256, key = u >> 6, du = u & 63;
          v8bf vv = *(const v8bf*)(vmat + (size_t)(mt + key) * EE + du * 8);
          int hh = du >> 3, ab = (du & 7) * 8;
#pragma unroll
          for (int jj = 0; jj < 8; ++jj) kv[hh][(ab + jj) * 32 + key] = vv[jj];
        }
        // f = mean over heads of p
#pragma unroll
        for (int e = t; e < 512; e += 256) {
          int r = e >> 5, c = e & 31;
          float fv = 0.0f;
#pragma unroll
          for (int hh = 0; hh < HH; ++hh) fv += (float)ps[hh][r * 32 + c];
          fout[(size_t)(b0 + r) * MK + mt + c] = fv * 0.125f;
        }
        __syncthreads();  // Vt staged
        // ctx[16q x 64a] += p[16q x 32m] @ v[32m x 64a]
        v16bf ap = load_frag(&ps[h][0], l15, 32);
#pragma unroll
        for (int nt = 0; nt < 4; ++nt) {
          v16bf bv = load_frag(&kv[h][0], nt * 16 + l15, 32);
          acc[nt] = WMMA_BF16(ap, bv, acc[nt]);
        }
      }
    }
    if (pass == 0) {
#pragma unroll
      for (int off = 8; off >= 1; off >>= 1)
#pragma unroll
        for (int i = 0; i < 8; ++i) {
          float om = __shfl_xor(mx[i], off, 32);
          float os = __shfl_xor(sm[i], off, 32);
          float nm = fmaxf(mx[i], om);
          sm[i] = sm[i] * __expf(mx[i] - nm) + os * __expf(om - nm);
          mx[i] = nm;
        }
#pragma unroll
      for (int i = 0; i < 8; ++i) rs[i] = 1.0f / sm[i];
    }
  }

#pragma unroll
  for (int nt = 0; nt < 4; ++nt)
#pragma unroll
    for (int i = 0; i < 8; ++i) {
      size_t row = b0 + i + ((lane >> 4) << 3);
      ctx[row * EE + h * 64 + nt * 16 + l15] = (__bf16)acc[nt][i];
    }
}

// ---------------------------------------------------------------------------
extern "C" void kernel_launch(void* const* d_in, const int* in_sizes, int n_in,
                              void* d_out, int out_size, void* d_ws, size_t ws_size,
                              hipStream_t stream) {
  const float* x   = (const float*)d_in[0];
  const float* Xc  = (const float*)d_in[1];
  const float* Wq  = (const float*)d_in[2];
  const float* Wk  = (const float*)d_in[3];
  const float* Wv  = (const float*)d_in[4];
  const float* qw  = (const float*)d_in[5];
  const float* kw  = (const float*)d_in[6];
  const float* vw  = (const float*)d_in[7];
  const float* b_q = (const float*)d_in[8];
  const float* b_k = (const float*)d_in[9];
  const float* b_v = (const float*)d_in[10];
  const float* ow  = (const float*)d_in[11];
  const float* ob  = (const float*)d_in[12];
  const float* Wo  = (const float*)d_in[13];
  const float* Wob = (const float*)d_in[14];

  float* out_x    = (float*)d_out;
  float* out_xhat = out_x + (size_t)BQ * NN;
  float* out_f    = out_xhat + (size_t)BQ * NN;
  float* out_V    = out_f + (size_t)BQ * MK;

  // bf16 workspace bump allocator (~64 MB total)
  __bf16* wp = (__bf16*)d_ws;
  auto alloc = [&](size_t n) { __bf16* p = wp; wp += n; return p; };
  __bf16* xb   = alloc((size_t)BQ * NN);
  __bf16* Xcb  = alloc((size_t)MK * NN);
  __bf16* Wqt  = alloc((size_t)EE * NN);   // [512,768]
  __bf16* Wkt  = alloc((size_t)EE * NN);
  __bf16* Wvt  = alloc((size_t)NN * NN);   // [768,768]
  __bf16* qwb  = alloc((size_t)EE * EE);
  __bf16* kwb  = alloc((size_t)EE * EE);
  __bf16* vwb  = alloc((size_t)EE * NN);
  __bf16* owb  = alloc((size_t)EE * EE);
  __bf16* Wo16 = alloc((size_t)NN * EE);
  __bf16* Qb   = alloc((size_t)BQ * EE);
  __bf16* Kb   = alloc((size_t)MK * EE);
  __bf16* Vb   = alloc((size_t)MK * EE);
  __bf16* qb   = alloc((size_t)BQ * EE);
  __bf16* kb   = alloc((size_t)MK * EE);
  __bf16* vb   = alloc((size_t)MK * EE);
  __bf16* ctxb = alloc((size_t)BQ * EE);
  __bf16* aob  = alloc((size_t)BQ * EE);

  dim3 blk(256);
  auto cvt = [&](const float* in, __bf16* o, size_t n) {
    cvt_kernel<<<dim3((unsigned)(n / 1024)), blk, 0, stream>>>(in, o, (int)(n / 4));
  };
  // one-time fp32 -> bf16 conversions
  cvt(x, xb, (size_t)BQ * NN);
  cvt(Xc, Xcb, (size_t)MK * NN);
  cvt(qw, qwb, (size_t)EE * EE);
  cvt(kw, kwb, (size_t)EE * EE);
  cvt(vw, vwb, (size_t)EE * NN);
  cvt(ow, owb, (size_t)EE * EE);
  cvt(Wo, Wo16, (size_t)NN * EE);
  // transpose-convert [K,N] -> [N,K] so every GEMM B operand is [N,K] bf16
  tcvt_kernel<<<dim3(EE / 256, NN), blk, 0, stream>>>(Wq, Wqt, NN, EE);
  tcvt_kernel<<<dim3(EE / 256, NN), blk, 0, stream>>>(Wk, Wkt, NN, EE);
  tcvt_kernel<<<dim3(NN / 256, NN), blk, 0, stream>>>(Wv, Wvt, NN, NN);

  // projections (grid = (N/64, M/128))
  gemm_bf16_kernel<<<dim3(EE / 64, BQ / 128), blk, 0, stream>>>(xb, Wqt, nullptr, nullptr, Qb, EE, NN, 0.125f);
  gemm_bf16_kernel<<<dim3(EE / 64, MK / 128), blk, 0, stream>>>(Xcb, Wkt, nullptr, nullptr, Kb, EE, NN, 1.0f);
  gemm_bf16_kernel<<<dim3(NN / 64, MK / 128), blk, 0, stream>>>(Xcb, Wvt, nullptr, out_V, Vb, NN, NN, 1.0f);
  gemm_bf16_kernel<<<dim3(EE / 64, BQ / 128), blk, 0, stream>>>(Qb, qwb, b_q, nullptr, qb, EE, EE, 1.0f);
  gemm_bf16_kernel<<<dim3(EE / 64, MK / 128), blk, 0, stream>>>(Kb, kwb, b_k, nullptr, kb, EE, EE, 1.0f);
  gemm_bf16_kernel<<<dim3(EE / 64, MK / 128), blk, 0, stream>>>(Vb, vwb, b_v, nullptr, vb, EE, NN, 1.0f);
  // attention core
  attention_kernel<<<dim3(BQ / 16), blk, 0, stream>>>(qb, kb, vb, ctxb, out_f);
  // output projections
  gemm_bf16_kernel<<<dim3(EE / 64, BQ / 128), blk, 0, stream>>>(ctxb, owb, ob, nullptr, aob, EE, EE, 1.0f);
  gemm_bf16_kernel<<<dim3(NN / 64, BQ / 128), blk, 0, stream>>>(aob, Wo16, Wob, out_xhat, nullptr, NN, EE, 1.0f);
  // x passthrough
  hipMemcpyAsync(out_x, x, (size_t)BQ * NN * sizeof(float), hipMemcpyDeviceToDevice, stream);
}
